// fTN_Transformer_Model_1391569404364
// MI455X (gfx1250) — matrix operand order repaired and checked by reading, not compiled, not run on hardware
//
#include <hip/hip_runtime.h>
#include <hip/hip_bf16.h>
#include <math.h>

// ---------------------------------------------------------------------------
// fTN transformer (enc-dec) for gfx1250.
// All GEMMs / attention matmuls via v_wmma_f32_16x16x32_f16 with f16-native
// operand staging (contiguous b128 loads), ping-pong software-pipelined K
// loop (no register-rotation movs), 64x32 C block per wave.
// B=32 S_SRC=64 S_TGT=1024 D=128 H=8 dh=16 DFF=512 LE=LD=6
// ---------------------------------------------------------------------------

typedef _Float16 f16;
typedef __attribute__((ext_vector_type(16))) _Float16 v16h;
typedef __attribute__((ext_vector_type(8)))  _Float16 v8h;
typedef __attribute__((ext_vector_type(8)))  float    v8f;

#define B_    32
#define S_SRC 64
#define S_TGT 1024
#define D_    128
#define H_    8
#define DH_   16
#define DFF_  512
#define LE_   6
#define LD_   6
#define MT_   4          // M tiles (16 rows each) per wave
#define NT_   2          // N tiles (16 cols each) per wave

static __device__ __forceinline__ v8f wmma_f16(v16h a, v16h b, v8f c) {
  // (neg_a, A, neg_b, B, c_mod, C, reuse_a, reuse_b)
  return __builtin_amdgcn_wmma_f32_16x16x32_f16(false, a, false, b, (short)0, c,
                                                false, false);
}

static __device__ __forceinline__ v16h cat16(v8h lo, v8h hi) {
  return __builtin_shufflevector(lo, hi, 0, 1, 2, 3, 4, 5, 6, 7,
                                         8, 9, 10, 11, 12, 13, 14, 15);
}

// PyTorch-style sinusoidal PE value for (pos, dim), d_model = 128
static __device__ __forceinline__ float pe_val(int pos, int d) {
  int i2 = d & ~1;
  float freq = __expf((float)i2 * (-9.210340371976184f / 128.0f)); // -ln(1e4)/D
  float ang = (float)pos * freq;
  return (d & 1) ? __cosf(ang) : __sinf(ang);
}

// f32 -> f16 elementwise (used once per launch for the weights)
__global__ void cast_kernel(const float* __restrict__ in, f16* __restrict__ out,
                            int n)
{
  int i = blockIdx.x * blockDim.x + threadIdx.x;
  if (i < n) out[i] = (f16)in[i];
}

// --------------------------- GEMM fragments --------------------------------
struct Frag {
  v8h  a0[MT_], a1[MT_];
  v16h b[NT_];
};

static __device__ __forceinline__ void load_frag(
    Frag& f, const f16* const* xrow, const f16* const* wrow, int k0)
{
#pragma unroll
  for (int t = 0; t < MT_; ++t) {
    f.a0[t] = *(const v8h*)(xrow[t] + k0);
    f.a1[t] = *(const v8h*)(xrow[t] + k0 + 8);
  }
#pragma unroll
  for (int n = 0; n < NT_; ++n) f.b[n] = *(const v16h*)(wrow[n] + k0);
}

static __device__ __forceinline__ void compute_frag(v8f acc[NT_][MT_],
                                                    const Frag& f)
{
#pragma unroll
  for (int n = 0; n < NT_; ++n)
#pragma unroll
    for (int t = 0; t < MT_; ++t)
      acc[n][t] = wmma_f16(cat16(f.a0[t], f.a1[t]), f.b[n], acc[n][t]);
}

// ---------------------------------------------------------------------------
// C[M,N] = act( X[M,K] @ W[N,K]^T + bias[N] );  X,W f16, C written as f32
// (Cf) and/or f16 (Ch).  One wave computes a 64x32 C block; the K loop is
// ping-pong double-buffered (2 chunks per iteration, alternating fragment
// buffers -> loads overlap WMMAs with no register-rotation copies).
// Fragment layouts used here (same k order in A and B) are contiguous f16 spans:
//   A lane L: elems 0..7  = X[m][k0 + 16*(L>>4) .. +7]
//             elems 8..15 = X[m][k0 + 16*(L>>4) + 8 .. +7]      (m = 16t+(L&15))
//   B lane L: elems 0..15 = W[n][k0 + 16*(L>>4) .. +15]         (n = 16u+(L&15))
//   C lane L, vgpr r -> (m = r + 8*(L>>4), n = L&15)
// Requires K multiple of 64 (chunk count even) -- true here (128, 512).
// ---------------------------------------------------------------------------
__global__ __launch_bounds__(32) void gemm_xwt_kernel(
    const f16* __restrict__ X, const f16* __restrict__ W,
    const float* __restrict__ bias,
    float* __restrict__ Cf, f16* __restrict__ Ch,
    int M, int N, int K, int relu)
{
  const int lane = threadIdx.x;
  const int lo = lane & 15, hi = lane >> 4;
  const int n0 = blockIdx.x * (16 * NT_);
  const int m0 = blockIdx.y * (16 * MT_);

  const f16* wrow[NT_];
#pragma unroll
  for (int n = 0; n < NT_; ++n)
    wrow[n] = W + (size_t)(n0 + n * 16 + lo) * K + hi * 16;
  const f16* xrow[MT_];
#pragma unroll
  for (int t = 0; t < MT_; ++t)
    xrow[t] = X + (size_t)(m0 + t * 16 + lo) * K + hi * 16;

  v8f acc[NT_][MT_] = {};
  Frag f0, f1;

  const int C = K >> 5;              // #32-wide chunks; even, >= 2
  load_frag(f0, xrow, wrow, 0);
  int c = 0;
  for (; c + 2 < C; c += 2) {
    __builtin_prefetch((const void*)(wrow[0] + ((c + 3) << 5)), 0, 1);
    load_frag(f1, xrow, wrow, (c + 1) << 5);
    compute_frag(acc, f0);           // chunk c
    load_frag(f0, xrow, wrow, (c + 2) << 5);
    compute_frag(acc, f1);           // chunk c+1
  }
  load_frag(f1, xrow, wrow, (C - 1) << 5);
  compute_frag(acc, f0);             // chunk C-2
  compute_frag(acc, f1);             // chunk C-1

#pragma unroll
  for (int n = 0; n < NT_; ++n) {
    float bv = bias[n0 + n * 16 + lo];
#pragma unroll
    for (int t = 0; t < MT_; ++t) {
#pragma unroll
      for (int r = 0; r < 8; ++r) {
        float val = acc[n][t][r] + bv;
        if (relu) val = fmaxf(val, 0.0f);
        size_t idx = (size_t)(m0 + t * 16 + 8 * hi + r) * N + n0 + n * 16 + lo;
        if (Cf) Cf[idx] = val;
        if (Ch) Ch[idx] = (f16)val;
      }
    }
  }
}

// Vt[b][h][dl][kpos] = QKV[(b*Sk + kpos)*ld + voff + h*16 + dl]
// so attention PV B-fragments become contiguous 32B loads.
__global__ void transpose_v_kernel(const f16* __restrict__ QKV, int ld, int voff,
                                   f16* __restrict__ Vt, int Sk)
{
  int idx = blockIdx.x * blockDim.x + threadIdx.x;
  int kpos = idx % Sk;
  int rest = idx / Sk;          // (b*H + h)*16 + dl
  int dl = rest & 15;
  int bh = rest >> 4;
  int h = bh % H_;
  int b = bh / H_;
  Vt[idx] = QKV[(size_t)(b * Sk + kpos) * ld + voff + h * 16 + dl];
}

// ---------------------------------------------------------------------------
// Fused attention: one wave per (b, head, 16-row q tile).
// scores = (Q @ K^T)*scale via WMMA (dh=16 zero-padded to K=32), f16 probs in
// 32KB LDS (P fragments are ds b128 loads), O = P @ V with pre-transposed V.
// ---------------------------------------------------------------------------
__global__ __launch_bounds__(32) void attn_kernel(
    const f16* __restrict__ Q, int ldq, int qoff,
    const f16* __restrict__ Kb, int ldk, int koff,
    const f16* __restrict__ Vt,      // [B,H,16,Sk]
    f16* __restrict__ O,             // [B*Sq, 128]
    int Sq, int Sk, float scale)
{
  __shared__ _Float16 sp[16 * 1024];         // 32KB scores/probs
  const int lane = threadIdx.x;
  const int lo = lane & 15, hi = lane >> 4;
  const int qt = blockIdx.x, h = blockIdx.y, b = blockIdx.z;
  const int bq = b * Sq + qt * 16;
  const int bk = b * Sk;

  // Q fragment: dh=16 real (contiguous 8 halves per lane), K=16..31 zero-pad.
  v8h z8 = {};
  v8h q0 = *(const v8h*)(Q + (size_t)(bq + lo) * ldq + qoff + h * 16 + hi * 8);
  v16h aq = cat16(q0, z8);

  // ---- scores into LDS (f16) ----
  for (int kt = 0; kt < Sk; kt += 16) {
    v16h bfk = cat16(*(const v8h*)(Kb + (size_t)(bk + kt + lo) * ldk + koff + h * 16 + hi * 8), z8);
    v8f s = {};
    s = wmma_f16(aq, bfk, s);
#pragma unroll
    for (int r = 0; r < 8; ++r)
      sp[(8 * hi + r) * Sk + kt + lo] = (f16)(s[r] * scale);
  }
  __syncthreads();

  // ---- softmax over Sk per row (whole wave per row) ----
  for (int r = 0; r < 16; ++r) {
    float mx = -3.4e38f;
    for (int j = lane; j < Sk; j += 32) mx = fmaxf(mx, (float)sp[r * Sk + j]);
#pragma unroll
    for (int o = 16; o > 0; o >>= 1) mx = fmaxf(mx, __shfl_xor(mx, o, 32));
    float sum = 0.0f;
    for (int j = lane; j < Sk; j += 32) {
      float e = __expf((float)sp[r * Sk + j] - mx);
      sp[r * Sk + j] = (f16)e;               // e <= 1, safe in f16
      sum += e;
    }
#pragma unroll
    for (int o = 16; o > 0; o >>= 1) sum += __shfl_xor(sum, o, 32);
    float inv = 1.0f / sum;
    for (int j = lane; j < Sk; j += 32)
      sp[r * Sk + j] = (f16)((float)sp[r * Sk + j] * inv);
  }
  __syncthreads();

  // ---- O = P @ V  (V pre-transposed: contiguous B fragments) ----
  const f16* vrow = Vt + ((size_t)(b * H_ + h) * 16 + lo) * Sk;
  v8f acc = {};
#pragma unroll 2
  for (int kt = 0; kt < Sk; kt += 32) {
    v8h p0 = *(const v8h*)(sp + lo * Sk + kt + hi * 16);
    v8h p1 = *(const v8h*)(sp + lo * Sk + kt + hi * 16 + 8);
    v16h bv = *(const v16h*)(vrow + kt + hi * 16);
    acc = wmma_f16(cat16(p0, p1), bv, acc);
  }
#pragma unroll
  for (int r = 0; r < 8; ++r)
    O[(size_t)(bq + 8 * hi + r) * D_ + h * 16 + lo] = (f16)acc[r];
}

// ---------------------------------------------------------------------------
// Y = LayerNorm(X + R)*g + b over D=128; writes f32 (Yf) and/or f16 (Yh).
// One wave per row, 4 elements per lane.
// ---------------------------------------------------------------------------
__global__ __launch_bounds__(32) void add_ln_kernel(
    const float* __restrict__ X, const float* __restrict__ R,
    const float* __restrict__ g, const float* __restrict__ beta,
    float* __restrict__ Yf, f16* __restrict__ Yh)
{
  const int row = blockIdx.x, lane = threadIdx.x;
  float v[4];
  float s = 0.0f;
#pragma unroll
  for (int i = 0; i < 4; ++i) {
    int c = lane + 32 * i;
    float x = X[(size_t)row * D_ + c];
    if (R) x += R[(size_t)row * D_ + c];
    v[i] = x;
    s += x;
  }
#pragma unroll
  for (int o = 16; o > 0; o >>= 1) s += __shfl_xor(s, o, 32);
  float mean = s * (1.0f / 128.0f);
  float var = 0.0f;
#pragma unroll
  for (int i = 0; i < 4; ++i) { float d = v[i] - mean; var += d * d; }
#pragma unroll
  for (int o = 16; o > 0; o >>= 1) var += __shfl_xor(var, o, 32);
  var *= (1.0f / 128.0f);
  float inv = rsqrtf(var + 1e-5f);
#pragma unroll
  for (int i = 0; i < 4; ++i) {
    int c = lane + 32 * i;
    float y = (v[i] - mean) * inv * g[c] + beta[c];
    if (Yf) Yf[(size_t)row * D_ + c] = y;
    if (Yh) Yh[(size_t)row * D_ + c] = (f16)y;
  }
}

// s[row,d] = emb[src[row]][d]*sqrt(D) + pe(row % S_SRC, d)   (f32 + f16)
__global__ void embed_src_kernel(const int* __restrict__ src,
                                 const float* __restrict__ emb,
                                 float* __restrict__ Sf, f16* __restrict__ Sh)
{
  int row = blockIdx.x, d = threadIdx.x;
  int pos = row % S_SRC;
  int tok = src[row];
  float y = emb[(size_t)tok * D_ + d] * 11.313708498984761f + pe_val(pos, d);
  Sf[(size_t)row * D_ + d] = y;
  Sh[(size_t)row * D_ + d] = (f16)y;
}

// t[row,d] = tgt[row,d] + pe(row % S_TGT, d)   (f32 + f16)
__global__ void init_tgt_kernel(const float* __restrict__ tgt,
                                float* __restrict__ Tf, f16* __restrict__ Th)
{
  int row = blockIdx.x, d = threadIdx.x;
  int pos = row % S_TGT;
  float y = tgt[(size_t)row * D_ + d] + pe_val(pos, d);
  Tf[(size_t)row * D_ + d] = y;
  Th[(size_t)row * D_ + d] = (f16)y;
}

// out[row] = dot(X[row,:], w) + b   (N=1 head -> wave reduction, fp32)
__global__ __launch_bounds__(32) void fc_kernel(const float* __restrict__ X,
                                                const float* __restrict__ w,
                                                const float* __restrict__ b,
                                                float* __restrict__ out)
{
  int row = blockIdx.x, lane = threadIdx.x;
  float s = 0.0f;
#pragma unroll
  for (int i = 0; i < 4; ++i)
    s += X[(size_t)row * D_ + lane + 32 * i] * w[lane + 32 * i];
#pragma unroll
  for (int o = 16; o > 0; o >>= 1) s += __shfl_xor(s, o, 32);
  if (lane == 0) out[row] = s + b[0];
}

// ---------------------------------------------------------------------------
extern "C" void kernel_launch(void* const* d_in, const int* in_sizes, int n_in,
                              void* d_out, int out_size, void* d_ws, size_t ws_size,
                              hipStream_t stream)
{
  (void)in_sizes; (void)n_in; (void)out_size; (void)ws_size;
  const int*   src          = (const int*)  d_in[0];
  const float* tgt          = (const float*)d_in[1];
  const float* emb          = (const float*)d_in[2];
  const float* enc_qkv_w    = (const float*)d_in[3];
  const float* enc_qkv_b    = (const float*)d_in[4];
  const float* enc_out_w    = (const float*)d_in[5];
  const float* enc_out_b    = (const float*)d_in[6];
  const float* enc_ln1_g    = (const float*)d_in[7];
  const float* enc_ln1_b    = (const float*)d_in[8];
  const float* enc_ff1_w    = (const float*)d_in[9];
  const float* enc_ff1_b    = (const float*)d_in[10];
  const float* enc_ff2_w    = (const float*)d_in[11];
  const float* enc_ff2_b    = (const float*)d_in[12];
  const float* enc_ln2_g    = (const float*)d_in[13];
  const float* enc_ln2_b    = (const float*)d_in[14];
  const float* enc_norm_g   = (const float*)d_in[15];
  const float* enc_norm_b   = (const float*)d_in[16];
  const float* dec_sa_qkv_w = (const float*)d_in[17];
  const float* dec_sa_qkv_b = (const float*)d_in[18];
  const float* dec_sa_out_w = (const float*)d_in[19];
  const float* dec_sa_out_b = (const float*)d_in[20];
  const float* dec_ln1_g    = (const float*)d_in[21];
  const float* dec_ln1_b    = (const float*)d_in[22];
  const float* dec_ca_qkv_w = (const float*)d_in[23];
  const float* dec_ca_qkv_b = (const float*)d_in[24];
  const float* dec_ca_out_w = (const float*)d_in[25];
  const float* dec_ca_out_b = (const float*)d_in[26];
  const float* dec_ln2_g    = (const float*)d_in[27];
  const float* dec_ln2_b    = (const float*)d_in[28];
  const float* dec_ff1_w    = (const float*)d_in[29];
  const float* dec_ff1_b    = (const float*)d_in[30];
  const float* dec_ff2_w    = (const float*)d_in[31];
  const float* dec_ff2_b    = (const float*)d_in[32];
  const float* dec_ln3_g    = (const float*)d_in[33];
  const float* dec_ln3_b    = (const float*)d_in[34];
  const float* dec_norm_g   = (const float*)d_in[35];
  const float* dec_norm_b   = (const float*)d_in[36];
  const float* fc_w         = (const float*)d_in[37];
  const float* fc_b         = (const float*)d_in[38];
  float* out = (float*)d_out;

  const int Ms = B_ * S_SRC;   // 2048
  const int Mt = B_ * S_TGT;   // 32768
  const float scale = 0.25f;   // 1/sqrt(dh)

  // ---- workspace carve-up: fp32 region, then f16 region (~107 MB total) ----
  float* fp = (float*)d_ws;
  size_t fo = 0;
  float* sact  = fp + fo; fo += (size_t)Ms * 128;   // enc residual stream
  float* sproj = fp + fo; fo += (size_t)Ms * 128;   // enc sublayer out
  float* tact  = fp + fo; fo += (size_t)Mt * 128;   // dec residual stream
  float* tproj = fp + fo; fo += (size_t)Mt * 128;   // dec sublayer out

  f16* hp = (f16*)(fp + fo);
  size_t ho = 0;
  f16* sact16  = hp + ho; ho += (size_t)Ms * 128;
  f16* sbig16  = hp + ho; ho += (size_t)Ms * 512;   // enc qkv / ffn hidden
  f16* sattn16 = hp + ho; ho += (size_t)Ms * 128;
  f16* svt     = hp + ho; ho += (size_t)Ms * 128;   // enc self Vt
  f16* mem16   = hp + ho; ho += (size_t)Ms * 128;
  f16* memkv16 = hp + ho; ho += (size_t)Ms * 256;   // cross K,V
  f16* cavt    = hp + ho; ho += (size_t)Ms * 128;   // cross Vt
  f16* tact16  = hp + ho; ho += (size_t)Mt * 128;
  f16* tbig16  = hp + ho; ho += (size_t)Mt * 512;   // dec qkv / q / hidden
  f16* tattn16 = hp + ho; ho += (size_t)Mt * 128;
  f16* tvt     = hp + ho; ho += (size_t)Mt * 128;   // dec self Vt
  // f16 weights
  f16* w_e_qkv = hp + ho; ho += (size_t)LE_ * 3 * D_ * D_;
  f16* w_e_out = hp + ho; ho += (size_t)LE_ * D_ * D_;
  f16* w_e_ff1 = hp + ho; ho += (size_t)LE_ * DFF_ * D_;
  f16* w_e_ff2 = hp + ho; ho += (size_t)LE_ * D_ * DFF_;
  f16* w_d_sqkv = hp + ho; ho += (size_t)LD_ * 3 * D_ * D_;
  f16* w_d_sout = hp + ho; ho += (size_t)LD_ * D_ * D_;
  f16* w_d_cqkv = hp + ho; ho += (size_t)LD_ * 3 * D_ * D_;
  f16* w_d_cout = hp + ho; ho += (size_t)LD_ * D_ * D_;
  f16* w_d_ff1 = hp + ho; ho += (size_t)LD_ * DFF_ * D_;
  f16* w_d_ff2 = hp + ho; ho += (size_t)LD_ * D_ * DFF_;

  auto cast = [&](const float* in, f16* o, int n) {
    cast_kernel<<<(n + 255) / 256, 256, 0, stream>>>(in, o, n);
  };
  cast(enc_qkv_w, w_e_qkv, LE_ * 3 * D_ * D_);
  cast(enc_out_w, w_e_out, LE_ * D_ * D_);
  cast(enc_ff1_w, w_e_ff1, LE_ * DFF_ * D_);
  cast(enc_ff2_w, w_e_ff2, LE_ * D_ * DFF_);
  cast(dec_sa_qkv_w, w_d_sqkv, LD_ * 3 * D_ * D_);
  cast(dec_sa_out_w, w_d_sout, LD_ * D_ * D_);
  cast(dec_ca_qkv_w, w_d_cqkv, LD_ * 3 * D_ * D_);
  cast(dec_ca_out_w, w_d_cout, LD_ * D_ * D_);
  cast(dec_ff1_w, w_d_ff1, LD_ * DFF_ * D_);
  cast(dec_ff2_w, w_d_ff2, LD_ * D_ * DFF_);

  auto gemm = [&](const f16* X, const f16* W, const float* bias,
                  float* Cf, f16* Ch, int M, int N, int K, int relu) {
    gemm_xwt_kernel<<<dim3(N / (16 * NT_), M / (16 * MT_)), 32, 0, stream>>>(
        X, W, bias, Cf, Ch, M, N, K, relu);
  };

  // ---------------- encoder ----------------
  embed_src_kernel<<<Ms, D_, 0, stream>>>(src, emb, sact, sact16);
  for (int i = 0; i < LE_; ++i) {
    gemm(sact16, w_e_qkv + (size_t)i * 3 * D_ * D_, enc_qkv_b + (size_t)i * 3 * D_,
         nullptr, sbig16, Ms, 3 * D_, D_, 0);
    transpose_v_kernel<<<(B_ * H_ * 16 * S_SRC) / 256, 256, 0, stream>>>(
        sbig16, 3 * D_, 2 * D_, svt, S_SRC);
    attn_kernel<<<dim3(S_SRC / 16, H_, B_), 32, 0, stream>>>(
        sbig16, 3 * D_, 0, sbig16, 3 * D_, D_, svt, sattn16, S_SRC, S_SRC, scale);
    gemm(sattn16, w_e_out + (size_t)i * D_ * D_, enc_out_b + (size_t)i * D_,
         sproj, nullptr, Ms, D_, D_, 0);
    add_ln_kernel<<<Ms, 32, 0, stream>>>(sact, sproj,
        enc_ln1_g + (size_t)i * D_, enc_ln1_b + (size_t)i * D_, sact, sact16);
    gemm(sact16, w_e_ff1 + (size_t)i * DFF_ * D_, enc_ff1_b + (size_t)i * DFF_,
         nullptr, sbig16, Ms, DFF_, D_, 1);
    gemm(sbig16, w_e_ff2 + (size_t)i * D_ * DFF_, enc_ff2_b + (size_t)i * D_,
         sproj, nullptr, Ms, D_, DFF_, 0);
    add_ln_kernel<<<Ms, 32, 0, stream>>>(sact, sproj,
        enc_ln2_g + (size_t)i * D_, enc_ln2_b + (size_t)i * D_, sact, sact16);
  }
  add_ln_kernel<<<Ms, 32, 0, stream>>>(sact, nullptr, enc_norm_g, enc_norm_b,
                                       nullptr, mem16);

  // ---------------- decoder ----------------
  init_tgt_kernel<<<Mt, D_, 0, stream>>>(tgt, tact, tact16);
  for (int i = 0; i < LD_; ++i) {
    // self-attention
    gemm(tact16, w_d_sqkv + (size_t)i * 3 * D_ * D_, dec_sa_qkv_b + (size_t)i * 3 * D_,
         nullptr, tbig16, Mt, 3 * D_, D_, 0);
    transpose_v_kernel<<<(B_ * H_ * 16 * S_TGT) / 256, 256, 0, stream>>>(
        tbig16, 3 * D_, 2 * D_, tvt, S_TGT);
    attn_kernel<<<dim3(S_TGT / 16, H_, B_), 32, 0, stream>>>(
        tbig16, 3 * D_, 0, tbig16, 3 * D_, D_, tvt, tattn16, S_TGT, S_TGT, scale);
    gemm(tattn16, w_d_sout + (size_t)i * D_ * D_, dec_sa_out_b + (size_t)i * D_,
         tproj, nullptr, Mt, D_, D_, 0);
    add_ln_kernel<<<Mt, 32, 0, stream>>>(tact, tproj,
        dec_ln1_g + (size_t)i * D_, dec_ln1_b + (size_t)i * D_, tact, tact16);
    // cross-attention: q from t (W rows 0..D-1), k/v from memory (rows D..3D-1)
    gemm(tact16, w_d_cqkv + (size_t)i * 3 * D_ * D_, dec_ca_qkv_b + (size_t)i * 3 * D_,
         nullptr, tbig16, Mt, D_, D_, 0);
    gemm(mem16, w_d_cqkv + (size_t)i * 3 * D_ * D_ + (size_t)D_ * D_,
         dec_ca_qkv_b + (size_t)i * 3 * D_ + D_, nullptr, memkv16, Ms, 2 * D_, D_, 0);
    transpose_v_kernel<<<(B_ * H_ * 16 * S_SRC) / 256, 256, 0, stream>>>(
        memkv16, 2 * D_, D_, cavt, S_SRC);
    attn_kernel<<<dim3(S_TGT / 16, H_, B_), 32, 0, stream>>>(
        tbig16, D_, 0, memkv16, 2 * D_, 0, cavt, tattn16, S_TGT, S_SRC, scale);
    gemm(tattn16, w_d_cout + (size_t)i * D_ * D_, dec_ca_out_b + (size_t)i * D_,
         tproj, nullptr, Mt, D_, D_, 0);
    add_ln_kernel<<<Mt, 32, 0, stream>>>(tact, tproj,
        dec_ln2_g + (size_t)i * D_, dec_ln2_b + (size_t)i * D_, tact, tact16);
    // ffn
    gemm(tact16, w_d_ff1 + (size_t)i * DFF_ * D_, dec_ff1_b + (size_t)i * DFF_,
         nullptr, tbig16, Mt, DFF_, D_, 1);
    gemm(tbig16, w_d_ff2 + (size_t)i * D_ * DFF_, dec_ff2_b + (size_t)i * D_,
         tproj, nullptr, Mt, D_, DFF_, 0);
    add_ln_kernel<<<Mt, 32, 0, stream>>>(tact, tproj,
        dec_ln3_g + (size_t)i * D_, dec_ln3_b + (size_t)i * D_, tact, tact16);
  }
  add_ln_kernel<<<Mt, 32, 0, stream>>>(tact, nullptr, dec_norm_g, dec_norm_b,
                                       tact, nullptr);
  fc_kernel<<<Mt, 32, 0, stream>>>(tact, fc_w, fc_b, out);
}
